// LocalAttentionLayer_48223892799522
// MI455X (gfx1250) — matrix operand order repaired and checked by reading, not compiled, not run on hardware
//
#include <hip/hip_runtime.h>
#include <hip/hip_bf16.h>

typedef __attribute__((ext_vector_type(16))) _Float16 v16h;
typedef __attribute__((ext_vector_type(8)))  _Float16 v8h;
typedef __attribute__((ext_vector_type(8)))  float    v8f;
typedef __attribute__((ext_vector_type(4)))  unsigned int v4u;
typedef __attribute__((ext_vector_type(8)))  int      v8i_t;
typedef __attribute__((ext_vector_type(4)))  int      v4i_t;

#define D_MODEL 512
#define SEQ     2048
#define NROWS   8192   // B*S = 4*2048
#define NHEAD   8
#define DH      64

#define KC        128            // K-chunk staged per TDM transfer (halfs)
#define LDS_PITCH 136            // 128 halfs + 8 halfs pad (272B = 68 banks)

// ---------------------------------------------------------------------------
// fp32 -> f16 conversion (weights)
// ---------------------------------------------------------------------------
__global__ __launch_bounds__(256) void cvt_f32_f16(const float* __restrict__ in,
                                                   _Float16* __restrict__ out, int n) {
    int i = blockIdx.x * 256 + threadIdx.x;
    if (i < n) out[i] = (_Float16)in[i];
}

// ---------------------------------------------------------------------------
// LayerNorm over D=512, one wave per row, f16 output for WMMA consumption
// ---------------------------------------------------------------------------
__global__ __launch_bounds__(256) void ln_f16(const float* __restrict__ x,
                                              const float* __restrict__ g,
                                              const float* __restrict__ b,
                                              _Float16* __restrict__ out) {
    int wave = threadIdx.x >> 5;
    int lane = threadIdx.x & 31;
    int row  = blockIdx.x * 8 + wave;
    const float* xr = x + (size_t)row * D_MODEL;

    float4 v[4];
    float sum = 0.f, sumsq = 0.f;
#pragma unroll
    for (int i = 0; i < 4; ++i) {
        v[i] = ((const float4*)xr)[lane + 32 * i];
        sum   += v[i].x + v[i].y + v[i].z + v[i].w;
        sumsq += v[i].x * v[i].x + v[i].y * v[i].y + v[i].z * v[i].z + v[i].w * v[i].w;
    }
#pragma unroll
    for (int off = 16; off > 0; off >>= 1) {
        sum   += __shfl_xor(sum, off, 32);
        sumsq += __shfl_xor(sumsq, off, 32);
    }
    float mu  = sum * (1.0f / D_MODEL);
    float var = sumsq * (1.0f / D_MODEL) - mu * mu;
    float rs  = rsqrtf(var + 1e-5f);

    _Float16* orow = out + (size_t)row * D_MODEL;
#pragma unroll
    for (int i = 0; i < 4; ++i) {
        int base = 4 * (lane + 32 * i);
        orow[base + 0] = (_Float16)((v[i].x - mu) * rs * g[base + 0] + b[base + 0]);
        orow[base + 1] = (_Float16)((v[i].y - mu) * rs * g[base + 1] + b[base + 1]);
        orow[base + 2] = (_Float16)((v[i].z - mu) * rs * g[base + 2] + b[base + 2]);
        orow[base + 3] = (_Float16)((v[i].w - mu) * rs * g[base + 3] + b[base + 3]);
    }
}

// ---------------------------------------------------------------------------
// TDM: DMA a 64-row x KC-col f16 tile of W (row stride K halfs) into LDS,
// padding +4 DWORDs after every 64 DWORDs (one row) -> LDS pitch 272 bytes.
// D# layout per CDNA5 ISA ch.8. Issued by one wave; tracked with TENSORcnt.
// ---------------------------------------------------------------------------
__device__ __forceinline__ void tdm_load_tile(const _Float16* gsrc, unsigned int lds_addr, int K) {
    unsigned long long ga = (unsigned long long)(uintptr_t)gsrc;
    v4u g0;
    g0[0] = 1u;                                   // count=1 (valid), user descriptor
    g0[1] = lds_addr;                             // LDS byte address
    g0[2] = (unsigned int)(ga & 0xFFFFFFFFu);     // global_addr[31:0]
    g0[3] = (unsigned int)((ga >> 32) & 0x1FFFFFFu) | (2u << 30);  // addr[56:32], type=2

    v8i_t g1;
    // word0: data_size=1 (2B) <<16 | pad_enable<<20 | pad_interval=5 (64 dw) <<22
    //        | pad_amount=3 (4 dw) <<25 ; workgroup_mask=0
    g1[0] = (int)((1u << 16) | (1u << 20) | (5u << 22) | (3u << 25));
    unsigned int td0 = KC;    // tensor_dim0 == tile_dim0 (no OOB clamp needed)
    unsigned int td1 = 64u;   // tensor_dim1 == tile_dim1
    g1[1] = (int)(td0 << 16);                       // bits 63:48 = tensor_dim0[15:0]
    g1[2] = (int)((td0 >> 16) | (td1 << 16));       // tensor_dim0[31:16] | tensor_dim1[15:0]
    g1[3] = (int)((td1 >> 16) | ((unsigned)KC << 16)); // tensor_dim1[31:16] | tile_dim0
    g1[4] = (int)(64u);                             // tile_dim1=64 | tile_dim2=0
    g1[5] = (int)(unsigned)K;                       // tensor_dim0_stride[31:0] = K
    g1[6] = 0;                                      // stride[47:32]=0 | dim1_stride[15:0]=0
    g1[7] = 0;

    v4i_t z4 = {0, 0, 0, 0};
    v8i_t z8 = {0, 0, 0, 0, 0, 0, 0, 0};
    __builtin_amdgcn_tensor_load_to_lds(g0, g1, z4, z4, z8, 0);
}

// ---------------------------------------------------------------------------
// Generic GEMM: out[n, m] = sum_k A16[n,k] * W16[m,k]  (+ bias, epilogue)
// A16: [N, K] f16 row-major (global, per-wave-unique rows);
// W16: [M, K] f16 row-major, staged 64xKC at a time into LDS via TDM,
//      double-buffered, shared by all 8 waves of the block.
// Wave tile 32x64 (2 A-frags x 4 B-frags = 8 WMMA / k-step);
// block = 8 waves -> 256x64. grid.x = M/64, grid.y = N/256.
// EPI: 0 = store f16, 1 = f32 + residual, 2 = f16 GELU, 3 = f32 + residual
// ---------------------------------------------------------------------------
template <int EPI>
__global__ __launch_bounds__(256) void gemm_wmma(const _Float16* __restrict__ A,
                                                 const _Float16* __restrict__ W,
                                                 const float* __restrict__ bias,
                                                 const float* __restrict__ resid,
                                                 void* __restrict__ out,
                                                 int K, int M) {
    __shared__ _Float16 bsm[2][64 * LDS_PITCH];

    const int wave = threadIdx.x >> 5;
    const int lane = threadIdx.x & 31;
    const int lh   = lane >> 4;
    const int lm   = lane & 15;

    const int r0 = blockIdx.y * 256 + wave * 32;  // activation-row tile base (32 rows)
    const int c0 = blockIdx.x * 64;               // feature-col tile base

    const _Float16* arow0 = A + (size_t)(r0 + lm) * K;        // rows r0..r0+15
    const _Float16* arow1 = A + (size_t)(r0 + 16 + lm) * K;   // rows r0+16..r0+31
    const int kbA = lh * 8;     // A: elems 0..7 at K[kb..], elems 8..15 at K[kb+16..]
    const int kbB = lh * 8;     // B: elems 0..7 at K[kb..], elems 8..15 at K[kb+16..]

    unsigned int lds_off[2];
    lds_off[0] = (unsigned int)(uintptr_t)(&bsm[0][0]);
    lds_off[1] = (unsigned int)(uintptr_t)(&bsm[1][0]);

    v8f acc[2][4];
#pragma unroll
    for (int i = 0; i < 2; ++i)
#pragma unroll
        for (int j = 0; j < 4; ++j) acc[i][j] = (v8f){};

    const int nch = K / KC;
    if (wave == 0)
        tdm_load_tile(W + (size_t)c0 * K, lds_off[0], K);

    for (int ch = 0; ch < nch; ++ch) {
        __builtin_amdgcn_s_wait_tensorcnt(0);   // no-op for waves with TENSORcnt==0
        __syncthreads();                        // staged chunk visible to all waves

        if (wave == 0 && ch + 1 < nch)
            tdm_load_tile(W + (size_t)c0 * K + (size_t)(ch + 1) * KC,
                          lds_off[(ch + 1) & 1], K);

        const _Float16* bbuf = &bsm[ch & 1][0];
        const int kg = ch * KC;  // global K base of this chunk
#pragma unroll
        for (int kk = 0; kk < KC; kk += 32) {
            union { v16h v; v8h h[2]; } a0, a1, b[4];
            a0.h[0] = *(const v8h*)(arow0 + kg + kk + kbA);
            a0.h[1] = *(const v8h*)(arow0 + kg + kk + kbA + 16);
            a1.h[0] = *(const v8h*)(arow1 + kg + kk + kbA);
            a1.h[1] = *(const v8h*)(arow1 + kg + kk + kbA + 16);

            const _Float16* bp = bbuf + (size_t)lm * LDS_PITCH + kk + kbB;
#pragma unroll
            for (int j = 0; j < 4; ++j) {
                b[j].h[0] = *(const v8h*)(bp + j * 16 * LDS_PITCH);
                b[j].h[1] = *(const v8h*)(bp + j * 16 * LDS_PITCH + 16);
            }
#pragma unroll
            for (int j = 0; j < 4; ++j) {
                acc[0][j] = __builtin_amdgcn_wmma_f32_16x16x32_f16(
                    false, a0.v, false, b[j].v, (short)0, acc[0][j], false, false);
                acc[1][j] = __builtin_amdgcn_wmma_f32_16x16x32_f16(
                    false, a1.v, false, b[j].v, (short)0, acc[1][j], false, false);
            }
        }
        __syncthreads();   // all waves done reading chunk before it is overwritten
    }

    // D layout: VGPR i, lanes 0-15 -> row i, lanes 16-31 -> row i+8; col = lane%16
#pragma unroll
    for (int g2 = 0; g2 < 2; ++g2) {
#pragma unroll
        for (int j = 0; j < 4; ++j) {
            const int col = c0 + j * 16 + lm;
            const float bv = bias[col];
#pragma unroll
            for (int i = 0; i < 8; ++i) {
                const int row = r0 + g2 * 16 + i + 8 * lh;
                const size_t idx = (size_t)row * M + col;
                float v = acc[g2][j][i] + bv;
                if (EPI == 0) {
                    ((_Float16*)out)[idx] = (_Float16)v;
                } else if (EPI == 1) {
                    ((float*)out)[idx] = v + resid[idx];
                } else if (EPI == 2) {
                    float gel = 0.5f * v * (1.0f + erff(v * 0.70710678118654752f));
                    ((_Float16*)out)[idx] = (_Float16)gel;
                } else {
                    ((float*)out)[idx] = v + resid[idx];
                }
            }
        }
    }
}

// ---------------------------------------------------------------------------
// Local windowed attention (radius 4 -> <=9 keys). One wave per (token, head).
// ---------------------------------------------------------------------------
__global__ __launch_bounds__(256) void attn_local(const _Float16* __restrict__ qkv,
                                                  _Float16* __restrict__ ctx) {
    const int wave = threadIdx.x >> 5;
    const int lane = threadIdx.x & 31;
    const int id = blockIdx.x * 8 + wave;
    const int n = id >> 3;
    const int h = id & 7;
    const int s = n & (SEQ - 1);

    const _Float16* qr = qkv + (size_t)n * 1536 + h * DH;
    const float q0 = (float)qr[lane];
    const float q1 = (float)qr[lane + 32];

    const int j0 = (s - 4 < 0 ? 0 : s - 4) - s;
    const int j1 = (s + 4 > SEQ - 1 ? SEQ - 1 : s + 4) - s;

    float p[9];
    float mx = -3.0e38f;
    for (int dj = j0; dj <= j1; ++dj) {
        const _Float16* kr = qkv + (size_t)(n + dj) * 1536 + D_MODEL + h * DH;
        float partial = q0 * (float)kr[lane] + q1 * (float)kr[lane + 32];
#pragma unroll
        for (int off = 16; off > 0; off >>= 1) partial += __shfl_xor(partial, off, 32);
        float sc = partial * 0.125f;
        p[dj - j0] = sc;
        mx = fmaxf(mx, sc);
    }
    float denom = 0.f;
    for (int dj = j0; dj <= j1; ++dj) {
        float e = __expf(p[dj - j0] - mx);
        p[dj - j0] = e;
        denom += e;
    }
    const float inv = 1.0f / denom;
    float c0 = 0.f, c1 = 0.f;
    for (int dj = j0; dj <= j1; ++dj) {
        const _Float16* vr = qkv + (size_t)(n + dj) * 1536 + 2 * D_MODEL + h * DH;
        float w = p[dj - j0] * inv;
        c0 += w * (float)vr[lane];
        c1 += w * (float)vr[lane + 32];
    }
    _Float16* orow = ctx + (size_t)n * D_MODEL + h * DH;
    orow[lane]      = (_Float16)c0;
    orow[lane + 32] = (_Float16)c1;
}

// ---------------------------------------------------------------------------
// Launch
// ---------------------------------------------------------------------------
extern "C" void kernel_launch(void* const* d_in, const int* in_sizes, int n_in,
                              void* d_out, int out_size, void* d_ws, size_t ws_size,
                              hipStream_t stream) {
    const float* x         = (const float*)d_in[0];
    const float* in_proj_w = (const float*)d_in[1];
    const float* in_proj_b = (const float*)d_in[2];
    const float* out_w     = (const float*)d_in[3];
    const float* out_b     = (const float*)d_in[4];
    const float* ff_w1     = (const float*)d_in[5];
    const float* ff_b1     = (const float*)d_in[6];
    const float* ff_w2     = (const float*)d_in[7];
    const float* ff_b2     = (const float*)d_in[8];
    const float* ln1_g     = (const float*)d_in[9];
    const float* ln1_b     = (const float*)d_in[10];
    const float* ln2_g     = (const float*)d_in[11];
    const float* ln2_b     = (const float*)d_in[12];

    char* ws = (char*)d_ws;
    size_t off = 0;
    auto take = [&](size_t bytes) { char* p = ws + off; off += (bytes + 255) & ~(size_t)255; return p; };

    _Float16* w_qkv16 = (_Float16*)take((size_t)1536 * 512 * 2);
    _Float16* w_out16 = (_Float16*)take((size_t)512 * 512 * 2);
    _Float16* w_ff116 = (_Float16*)take((size_t)2048 * 512 * 2);
    _Float16* w_ff216 = (_Float16*)take((size_t)512 * 2048 * 2);
    _Float16* h16     = (_Float16*)take((size_t)NROWS * 512 * 2);   // reused for LN2 output
    _Float16* qkv16   = (_Float16*)take((size_t)NROWS * 1536 * 2);
    _Float16* ctx16   = (_Float16*)take((size_t)NROWS * 512 * 2);
    float*    x1      = (float*)take((size_t)NROWS * 512 * 4);
    _Float16* ff116   = (_Float16*)take((size_t)NROWS * 2048 * 2);

    cvt_f32_f16<<<(1536 * 512 + 255) / 256, 256, 0, stream>>>(in_proj_w, w_qkv16, 1536 * 512);
    cvt_f32_f16<<<(512 * 512 + 255) / 256, 256, 0, stream>>>(out_w, w_out16, 512 * 512);
    cvt_f32_f16<<<(2048 * 512 + 255) / 256, 256, 0, stream>>>(ff_w1, w_ff116, 2048 * 512);
    cvt_f32_f16<<<(512 * 2048 + 255) / 256, 256, 0, stream>>>(ff_w2, w_ff216, 512 * 2048);

    ln_f16<<<NROWS / 8, 256, 0, stream>>>(x, ln1_g, ln1_b, h16);

    gemm_wmma<0><<<dim3(1536 / 64, NROWS / 256), 256, 0, stream>>>(
        h16, w_qkv16, in_proj_b, nullptr, qkv16, 512, 1536);

    attn_local<<<(NROWS * NHEAD) / 8, 256, 0, stream>>>(qkv16, ctx16);

    gemm_wmma<1><<<dim3(512 / 64, NROWS / 256), 256, 0, stream>>>(
        ctx16, w_out16, out_b, x, x1, 512, 512);

    ln_f16<<<NROWS / 8, 256, 0, stream>>>(x1, ln2_g, ln2_b, h16);

    gemm_wmma<2><<<dim3(2048 / 64, NROWS / 256), 256, 0, stream>>>(
        h16, w_ff116, ff_b1, nullptr, ff116, 512, 2048);

    gemm_wmma<3><<<dim3(512 / 64, NROWS / 256), 256, 0, stream>>>(
        ff116, w_ff216, ff_b2, x1, (float*)d_out, 2048, 512);
}
